// NodeModel_65335042507073
// MI455X (gfx1250) — matrix operand order repaired and checked by reading, not compiled, not run on hardware
//
#include <hip/hip_runtime.h>
#include <hip/hip_bf16.h>

typedef unsigned short u16;
typedef __attribute__((ext_vector_type(16))) __bf16 v16bf;
typedef __attribute__((ext_vector_type(8)))  float  v8f;

#define NN 50000
#define EE 800000
#define ND 64
#define ED 32
#define HH 128
#define BN_EPS 1e-5f

struct Frag32B { uint4 lo, hi; };   // 32 bytes = one v16bf
struct BB16 { u16 u[16]; };         // 32 bytes
struct BB8  { u16 u[8]; };          // 16 bytes

__device__ __forceinline__ u16 f2bf(float f) {
  unsigned b = __builtin_bit_cast(unsigned, f);
  unsigned r = b + 0x7FFFu + ((b >> 16) & 1u);   // RNE
  return (u16)(r >> 16);
}
__device__ __forceinline__ float bf2f(u16 u) {
  unsigned b = ((unsigned)u) << 16;
  return __builtin_bit_cast(float, b);
}
__device__ __forceinline__ uint2 pack4(float4 v) {
  uint2 r;
  r.x = (unsigned)f2bf(v.x) | ((unsigned)f2bf(v.y) << 16);
  r.y = (unsigned)f2bf(v.z) | ((unsigned)f2bf(v.w) << 16);
  return r;
}

// A-fragment (16x32 bf16, M x K): lane lr = row M; 16 contiguous K at half*16.
__device__ __forceinline__ v16bf frag_a(const u16* tile, int ld, int lr, int hf) {
  const u16* p = tile + lr * ld + hf * 16;
  Frag32B t; t.lo = *(const uint4*)p; t.hi = *(const uint4*)(p + 8);
  return __builtin_bit_cast(v16bf, t);
}
// B-fragment (32x16 bf16, K x N) from N-major storage: lane lr = col N; 16 contiguous K at half*16.
__device__ __forceinline__ v16bf frag_b(const u16* tile, int ld, int lr, int hf) {
  const u16* p = tile + lr * ld + hf * 16;
  Frag32B t; t.lo = *(const uint4*)p; t.hi = *(const uint4*)(p + 8);
  return __builtin_bit_cast(v16bf, t);
}
__device__ __forceinline__ v8f wmma_bf16(v16bf a, v16bf b, v8f c) {
  return __builtin_amdgcn_wmma_f32_16x16x32_bf16(false, a, false, b, (short)0, c, false, false);
}

// Vectorized LDS copy of bf16 weight tiles: 8-element (16B) chunks, padded dest stride.
__device__ __forceinline__ void stage_w(u16* dst, int dstride, const u16* __restrict__ src,
                                        int sstride, int rows, int tid) {
  const int nchunk = sstride >> 3;
  for (int i = tid; i < rows * nchunk; i += 256) {
    int f = i / nchunk, c = i - f * nchunk;
    *(uint4*)&dst[f * dstride + c * 8] = *(const uint4*)&src[f * sstride + c * 8];
  }
}

// Stage msg tile [128 edges][96] = concat(x[row], edge_attr): 2 threads/row, float4 gathers,
// packed bf16 ds_store_b128. Concat boundary (64) is float4-aligned.
__device__ __forceinline__ void stage_edge_rows(u16* sB, const float* __restrict__ x,
                                                const float* __restrict__ ea,
                                                const int* __restrict__ ei, int e0, int tid) {
  const int r = tid >> 1;
  const int sh = tid & 1;
  const int e = e0 + r;
  const float* src0 = x + (size_t)ei[e] * ND;   // cols 0..63
  const float* src1 = ea + (size_t)e * ED;      // cols 64..95
#pragma unroll
  for (int c3 = 0; c3 < 3; ++c3) {
#pragma unroll
    for (int q = 0; q < 2; ++q) {
      int cc = sh * 48 + c3 * 16 + q * 8;
      float4 va = (cc < ND) ? *(const float4*)(src0 + cc) : *(const float4*)(src1 + cc - ND);
      float4 vb = (cc + 4 < ND) ? *(const float4*)(src0 + cc + 4) : *(const float4*)(src1 + cc + 4 - ND);
      uint2 pa = pack4(va), pb = pack4(vb);
      uint4 o; o.x = pa.x; o.y = pa.y; o.z = pb.x; o.w = pb.y;
      *(uint4*)&sB[r * 104 + cc] = o;
    }
  }
}

// ---------------- prep: weights f32 -> bf16, transposed [out_feat][k] ----------------
__global__ void k_prep(const float* __restrict__ W1a, const float* __restrict__ W1b,
                       const float* __restrict__ W2a, const float* __restrict__ W2b,
                       u16* __restrict__ w1aT, u16* __restrict__ w1bT,
                       u16* __restrict__ w2aT, u16* __restrict__ w2bT) {
  int id = blockIdx.x * 256 + threadIdx.x;
  if (id < 128 * 96) { int f = id / 96, k = id - f * 96; w1aT[id] = f2bf(W1a[k * HH + f]); return; }
  id -= 128 * 96;
  if (id < 128 * 128) { int f = id >> 7, k = id & 127; w1bT[id] = f2bf(W1b[k * HH + f]); return; }
  id -= 128 * 128;
  if (id < 128 * 192) { int f = id / 192, k = id - f * 192; w2aT[id] = f2bf(W2a[k * HH + f]); return; }
  id -= 128 * 192;
  if (id < 64 * 128) { int f = id >> 7, k = id & 127; w2bT[id] = f2bf(W2b[k * ND + f]); return; }
}

__global__ void k_zero(float* __restrict__ a, long na, float* __restrict__ b, int nb) {
  long i = (long)blockIdx.x * blockDim.x + threadIdx.x;
  long stride = (long)gridDim.x * blockDim.x;
  for (; i < na; i += stride) a[i] = 0.f;
  if (blockIdx.x == 0)
    for (int j = threadIdx.x; j < nb; j += blockDim.x) b[j] = 0.f;
}

// finalize BN: coef[f]=a=g/sqrt(var+eps), coef[128+f]=be - mean*a   (Lin bias cancels in BN)
__global__ void k_finalize(const float* __restrict__ sum, const float* __restrict__ ssq,
                           const float* __restrict__ g, const float* __restrict__ be,
                           float inv_cnt, float* __restrict__ coef) {
  int t = threadIdx.x;  // 128
  float m = sum[t] * inv_cnt;
  float var = ssq[t] * inv_cnt - m * m;
  float a = g[t] * rsqrtf(var + BN_EPS);
  coef[t] = a;
  coef[128 + t] = be[t] - m * a;
}

// ---------------- pass 1: edge GEMM1, accumulate per-column sum/sumsq ----------------
__global__ __launch_bounds__(256) void k_edge_stats(
    const float* __restrict__ x, const int* __restrict__ ei,
    const float* __restrict__ ea, const u16* __restrict__ w1aT,
    float* __restrict__ gsum, float* __restrict__ gssq) {
  __shared__ u16 sB[128 * 104];
  __shared__ u16 sA[128 * 104];
  __shared__ float ssum[128];
  __shared__ float sssq[128];
  const int tid = threadIdx.x;
  const int e0 = blockIdx.x * 128;
  if (tid < 128) { ssum[tid] = 0.f; sssq[tid] = 0.f; }
  stage_w(sA, 104, w1aT, 96, 128, tid);
  stage_edge_rows(sB, x, ea, ei, e0, tid);
  __syncthreads();
  const int w = tid >> 5, lane = tid & 31, lr = lane & 15, hf = lane >> 4;
  v8f acc[8];
#pragma unroll
  for (int m = 0; m < 8; ++m) {
#pragma unroll
    for (int r = 0; r < 8; ++r) acc[m][r] = 0.f;
  }
#pragma unroll
  for (int kt = 0; kt < 3; ++kt) {
    v16bf b = frag_b(&sB[(16 * w) * 104 + 32 * kt], 104, lr, hf);
#pragma unroll
    for (int mt = 0; mt < 8; ++mt) {
      v16bf a = frag_a(&sA[(16 * mt) * 104 + 32 * kt], 104, lr, hf);
      acc[mt] = wmma_bf16(a, b, acc[mt]);
    }
  }
#pragma unroll
  for (int mt = 0; mt < 8; ++mt) {
#pragma unroll
    for (int r = 0; r < 8; ++r) {
      float s = acc[mt][r];
      float q = s * s;
#pragma unroll
      for (int m = 1; m < 16; m <<= 1) { s += __shfl_xor(s, m, 32); q += __shfl_xor(q, m, 32); }
      if (lr == 0) {
        atomicAdd(&ssum[16 * mt + hf * 8 + r], s);
        atomicAdd(&sssq[16 * mt + hf * 8 + r], q);
      }
    }
  }
  __syncthreads();
  if (tid < 128) unsafeAtomicAdd(&gsum[tid], ssum[tid]);
  else if (tid < 256) unsafeAtomicAdd(&gssq[tid - 128], sssq[tid - 128]);
}

// ---- pass 2: fused edge GEMM1 -> BN -> ReLU -> (register transpose) -> GEMM2 -> scatter ----
__global__ __launch_bounds__(256) void k_edge_mlp(
    const float* __restrict__ x, const int* __restrict__ ei,
    const float* __restrict__ ea, const u16* __restrict__ w1aT,
    const u16* __restrict__ w1bT, const float* __restrict__ b1b,
    const float* __restrict__ coef1, float* __restrict__ agg) {
  __shared__ u16 smem[2 * 128 * 104];
  __shared__ float sc[384];  // a1[128], c1'[128], b1b[128]
  const int tid = threadIdx.x;
  const int e0 = blockIdx.x * 128;
  u16* sB = smem;
  u16* sA1 = smem + 128 * 104;
  for (int i = tid; i < 384; i += 256) sc[i] = (i < 256) ? coef1[i] : b1b[i - 256];
  stage_w(sA1, 104, w1aT, 96, 128, tid);
  stage_edge_rows(sB, x, ea, ei, e0, tid);
  __syncthreads();
  const int w = tid >> 5, lane = tid & 31, lr = lane & 15, hf = lane >> 4;
  v8f acc1[8];
#pragma unroll
  for (int m = 0; m < 8; ++m) {
#pragma unroll
    for (int r = 0; r < 8; ++r) acc1[m][r] = 0.f;
  }
#pragma unroll
  for (int kt = 0; kt < 3; ++kt) {
    v16bf b = frag_b(&sB[(16 * w) * 104 + 32 * kt], 104, lr, hf);
#pragma unroll
    for (int mt = 0; mt < 8; ++mt) {
      v16bf a = frag_a(&sA1[(16 * mt) * 104 + 32 * kt], 104, lr, hf);
      acc1[mt] = wmma_bf16(a, b, acc1[mt]);
    }
  }
  __syncthreads();
  // restage W1bT [128][128] into smem (reuses msg/W1a space), padded stride 136
  u16* sA2 = smem;
  stage_w(sA2, 136, w1bT, 128, 128, tid);
  __syncthreads();
  v8f acc2[8];
#pragma unroll
  for (int m = 0; m < 8; ++m) {
#pragma unroll
    for (int r = 0; r < 8; ++r) acc2[m][r] = 0.f;
  }
#pragma unroll
  for (int kt = 0; kt < 4; ++kt) {
    // Build B-fragment of relu(a1*acc1 + c1') in registers: lane owns half the
    // features of tiles 2kt/2kt+1; swap the other half with lane^16.
    BB16 bb;
#pragma unroll
    for (int r = 0; r < 8; ++r) {
      int f0 = 32 * kt + hf * 8 + r;        // feature of acc1[2kt][r] on this lane
      int f1 = 32 * kt + 16 + hf * 8 + r;   // feature of acc1[2kt+1][r]
      float v0 = fmaxf(0.f, sc[f0] * acc1[2 * kt][r] + sc[128 + f0]);
      float v1 = fmaxf(0.f, sc[f1] * acc1[2 * kt + 1][r] + sc[128 + f1]);
      float p0 = __shfl_xor(v0, 16, 32);
      float p1 = __shfl_xor(v1, 16, 32);
      bb.u[r]     = f2bf(hf ? p1 : v0);
      bb.u[8 + r] = f2bf(hf ? v1 : p0);
    }
    v16bf b = __builtin_bit_cast(v16bf, bb);
#pragma unroll
    for (int mt = 0; mt < 8; ++mt) {
      v16bf a = frag_a(&sA2[(16 * mt) * 136 + 32 * kt], 136, lr, hf);
      acc2[mt] = wmma_bf16(a, b, acc2[mt]);
    }
  }
  // scatter-add msg into agg[col[e]]
  const int e = e0 + 16 * w + lr;
  const int dest = ei[EE + e];
#pragma unroll
  for (int mt = 0; mt < 8; ++mt) {
#pragma unroll
    for (int r = 0; r < 8; ++r) {
      int f = 16 * mt + hf * 8 + r;
      unsafeAtomicAdd(&agg[(size_t)dest * HH + f], acc2[mt][r] + sc[256 + f]);
    }
  }
}

// ---------------- pass 3: node GEMM1 (K=192 in two 96-chunks), stats + h3 store ----------------
__global__ __launch_bounds__(256) void k_node_h3(
    const float* __restrict__ x, const float* __restrict__ agg,
    const u16* __restrict__ w2aT, float* __restrict__ gsum,
    float* __restrict__ gssq, u16* __restrict__ h3) {
  __shared__ u16 sB[128 * 104];
  __shared__ u16 sA[128 * 104];
  __shared__ float ssum[128];
  __shared__ float sssq[128];
  const int tid = threadIdx.x;
  const int n0 = blockIdx.x * 128;
  if (tid < 128) { ssum[tid] = 0.f; sssq[tid] = 0.f; }
  const int w = tid >> 5, lane = tid & 31, lr = lane & 15, hf = lane >> 4;
  v8f acc[8];
#pragma unroll
  for (int m = 0; m < 8; ++m) {
#pragma unroll
    for (int r = 0; r < 8; ++r) acc[m][r] = 0.f;
  }
  const int rr = tid >> 1, sh = tid & 1;
  const int nodeR = n0 + rr;
  const bool okr = nodeR < NN;
  const float* nx = x + (size_t)nodeR * ND;
  const float* nagg = agg + (size_t)nodeR * HH;
  for (int kc = 0; kc < 2; ++kc) {
    __syncthreads();  // protect ssum init / previous chunk's reads
    for (int i = tid; i < 128 * 12; i += 256) {
      int f = i / 12, c = i - f * 12;
      *(uint4*)&sA[f * 104 + c * 8] = *(const uint4*)&w2aT[f * 192 + kc * 96 + c * 8];
    }
#pragma unroll
    for (int c3 = 0; c3 < 3; ++c3) {
#pragma unroll
      for (int q = 0; q < 2; ++q) {
        int j = sh * 48 + c3 * 16 + q * 8;    // col within this 96-chunk
        int gc = kc * 96 + j;                 // global concat col (0..191)
        float4 va = make_float4(0.f, 0.f, 0.f, 0.f), vb = va;
        if (okr) {
          va = (gc < ND) ? *(const float4*)(nx + gc) : *(const float4*)(nagg + gc - ND);
          vb = (gc + 4 < ND) ? *(const float4*)(nx + gc + 4) : *(const float4*)(nagg + gc + 4 - ND);
        }
        uint2 pa = pack4(va), pb = pack4(vb);
        uint4 o; o.x = pa.x; o.y = pa.y; o.z = pb.x; o.w = pb.y;
        *(uint4*)&sB[rr * 104 + j] = o;
      }
    }
    __syncthreads();
#pragma unroll
    for (int kt = 0; kt < 3; ++kt) {
      v16bf b = frag_b(&sB[(16 * w) * 104 + 32 * kt], 104, lr, hf);
#pragma unroll
      for (int mt = 0; mt < 8; ++mt) {
        v16bf a = frag_a(&sA[(16 * mt) * 104 + 32 * kt], 104, lr, hf);
        acc[mt] = wmma_bf16(a, b, acc[mt]);
      }
    }
  }
  const int node = n0 + 16 * w + lr;
  const bool ok = node < NN;
#pragma unroll
  for (int mt = 0; mt < 8; ++mt) {
    BB8 pack;
#pragma unroll
    for (int r = 0; r < 8; ++r) {
      pack.u[r] = f2bf(acc[mt][r]);
      float s = ok ? acc[mt][r] : 0.f;
      float q = s * s;
#pragma unroll
      for (int m = 1; m < 16; m <<= 1) { s += __shfl_xor(s, m, 32); q += __shfl_xor(q, m, 32); }
      if (lr == 0) {
        atomicAdd(&ssum[16 * mt + hf * 8 + r], s);
        atomicAdd(&sssq[16 * mt + hf * 8 + r], q);
      }
    }
    if (ok) *(uint4*)&h3[(size_t)node * HH + 16 * mt + hf * 8] = __builtin_bit_cast(uint4, pack);
  }
  __syncthreads();
  if (tid < 128) unsafeAtomicAdd(&gsum[tid], ssum[tid]);
  else if (tid < 256) unsafeAtomicAdd(&gssq[tid - 128], sssq[tid - 128]);
}

// ---------------- pass 4: node BN -> ReLU -> GEMM2 -> out [N,64] ----------------
__global__ __launch_bounds__(256) void k_node_out(
    const u16* __restrict__ h3, const u16* __restrict__ w2bT,
    const float* __restrict__ coef2, const float* __restrict__ b2b,
    float* __restrict__ out) {
  __shared__ u16 sA[64 * 136];
  __shared__ float sc[320];  // a2[128], c2'[128], b2b[64]
  const int tid = threadIdx.x;
  for (int i = tid; i < 320; i += 256) sc[i] = (i < 256) ? coef2[i] : b2b[i - 256];
  stage_w(sA, 136, w2bT, 128, 64, tid);
  __syncthreads();
  const int n0 = blockIdx.x * 128;
  const int w = tid >> 5, lane = tid & 31, lr = lane & 15, hf = lane >> 4;
  const int node = n0 + 16 * w + lr;
  const bool ok = node < NN;
  const int nclamp = ok ? node : (NN - 1);
  v8f acc[4];
#pragma unroll
  for (int m = 0; m < 4; ++m) {
#pragma unroll
    for (int r = 0; r < 8; ++r) acc[m][r] = 0.f;
  }
#pragma unroll
  for (int kt = 0; kt < 4; ++kt) {
    const u16* src = h3 + (size_t)nclamp * HH + 32 * kt + hf * 16;
    Frag32B raw; raw.lo = *(const uint4*)src; raw.hi = *(const uint4*)(src + 8);
    BB16 rb = __builtin_bit_cast(BB16, raw);
    BB16 bb;
#pragma unroll
    for (int i = 0; i < 16; ++i) {
      int f = 32 * kt + hf * 16 + i;
      bb.u[i] = f2bf(fmaxf(0.f, sc[f] * bf2f(rb.u[i]) + sc[128 + f]));
    }
    v16bf b = __builtin_bit_cast(v16bf, bb);
#pragma unroll
    for (int mt = 0; mt < 4; ++mt) {
      v16bf a = frag_a(&sA[(16 * mt) * 136 + 32 * kt], 136, lr, hf);
      acc[mt] = wmma_bf16(a, b, acc[mt]);
    }
  }
  if (ok) {
#pragma unroll
    for (int mt = 0; mt < 4; ++mt) {
      float4 f0, f1;
      f0.x = acc[mt][0] + sc[256 + 16 * mt + hf * 8 + 0];
      f0.y = acc[mt][1] + sc[256 + 16 * mt + hf * 8 + 1];
      f0.z = acc[mt][2] + sc[256 + 16 * mt + hf * 8 + 2];
      f0.w = acc[mt][3] + sc[256 + 16 * mt + hf * 8 + 3];
      f1.x = acc[mt][4] + sc[256 + 16 * mt + hf * 8 + 4];
      f1.y = acc[mt][5] + sc[256 + 16 * mt + hf * 8 + 5];
      f1.z = acc[mt][6] + sc[256 + 16 * mt + hf * 8 + 6];
      f1.w = acc[mt][7] + sc[256 + 16 * mt + hf * 8 + 7];
      *(float4*)&out[(size_t)node * ND + 16 * mt + hf * 8] = f0;
      *(float4*)&out[(size_t)node * ND + 16 * mt + hf * 8 + 4] = f1;
    }
  }
}

extern "C" void kernel_launch(void* const* d_in, const int* in_sizes, int n_in,
                              void* d_out, int out_size, void* d_ws, size_t ws_size,
                              hipStream_t stream) {
  (void)in_sizes; (void)n_in; (void)out_size; (void)ws_size;
  const float* x   = (const float*)d_in[0];
  const int*   ei  = (const int*)d_in[1];    // [2,E] flat: rows then cols
  const float* ea  = (const float*)d_in[2];
  const float* W1a = (const float*)d_in[5];
  const float* g1  = (const float*)d_in[7];
  const float* be1 = (const float*)d_in[8];
  const float* W1b = (const float*)d_in[9];
  const float* b1b = (const float*)d_in[10];
  const float* W2a = (const float*)d_in[11];
  const float* g2  = (const float*)d_in[13];
  const float* be2 = (const float*)d_in[14];
  const float* W2b = (const float*)d_in[15];
  const float* b2b = (const float*)d_in[16];

  char* p = (char*)d_ws;
  u16* w1aT = (u16*)p; p += 128 * 96 * 2;
  u16* w1bT = (u16*)p; p += 128 * 128 * 2;
  u16* w2aT = (u16*)p; p += 128 * 192 * 2;
  u16* w2bT = (u16*)p; p += 64 * 128 * 2;
  float* stats = (float*)p; p += 512 * 4;   // sum1,ssq1,sum2,ssq2
  float* coef1 = (float*)p; p += 256 * 4;
  float* coef2 = (float*)p; p += 256 * 4;
  float* agg = (float*)p; p += (size_t)NN * HH * 4;
  u16* h3 = (u16*)p; p += (size_t)50048 * HH * 2;

  k_prep<<<(61440 + 255) / 256, 256, 0, stream>>>(W1a, W1b, W2a, W2b, w1aT, w1bT, w2aT, w2bT);
  k_zero<<<2048, 256, 0, stream>>>(agg, (long)NN * HH, stats, 512);
  k_edge_stats<<<EE / 128, 256, 0, stream>>>(x, ei, ea, w1aT, stats, stats + 128);
  k_finalize<<<1, 128, 0, stream>>>(stats, stats + 128, g1, be1, 1.0f / (float)EE, coef1);
  k_edge_mlp<<<EE / 128, 256, 0, stream>>>(x, ei, ea, w1aT, w1bT, b1b, coef1, agg);
  k_node_h3<<<(NN + 127) / 128, 256, 0, stream>>>(x, agg, w2aT, stats + 256, stats + 384, h3);
  k_finalize<<<1, 128, 0, stream>>>(stats + 256, stats + 384, g2, be2, 1.0f / (float)NN, coef2);
  k_node_out<<<(NN + 127) / 128, 256, 0, stream>>>(h3, w2bT, coef2, b2b, (float*)d_out);
}
